// MultiHeadSelfAttention_13726715478412
// MI455X (gfx1250) — matrix operand order repaired and checked by reading, not compiled, not run on hardware
//
#include <hip/hip_runtime.h>
#include <hip/hip_bf16.h>

#define DIMK   1024
#define HEADS  16
#define DH     64
#define HD     1024
#define BATCH  2
#define TLEN   2048
#define MTOT   (BATCH*TLEN)   // 4096
#define SCALE  0.125f         // 1/sqrt(64)

typedef __attribute__((ext_vector_type(16))) __bf16 v16bf;
typedef __attribute__((ext_vector_type(8)))  float  v8f;

union Frag16 { uint4 q[2]; v16bf v; };

__device__ __forceinline__ unsigned short f2bf(float f) {
  return __builtin_bit_cast(unsigned short, (__bf16)f);   // RNE fptrunc -> native bf16 cvt
}
__device__ __forceinline__ unsigned pk2bf(float lo, float hi) {
  return (unsigned)f2bf(lo) | ((unsigned)f2bf(hi) << 16);
}

// A-fragment (16x32, 16-bit): per lane, K chunks [koff, koff+8) and [koff+16, koff+24)
__device__ __forceinline__ v16bf load_fragA(const unsigned short* p) {
  Frag16 f;
  f.q[0] = *(const uint4*)p;
  f.q[1] = *(const uint4*)(p + 16);
  return f.v;
}
// B-fragment (32x16, 16-bit): per lane, K chunks [koff, koff+8) and [koff+16, koff+24) of its column
__device__ __forceinline__ v16bf load_fragB(const unsigned short* p) {
  Frag16 f;
  f.q[0] = *(const uint4*)p;
  f.q[1] = *(const uint4*)(p + 16);
  return f.v;
}

__device__ __forceinline__ v8f wmma_bf16(v16bf a, v16bf b, v8f c) {
  return __builtin_amdgcn_wmma_f32_16x16x32_bf16(false, a, false, b, (short)0, c, false, false);
}

__device__ __forceinline__ v8f zero8() {
  v8f z;
#pragma unroll
  for (int i = 0; i < 8; ++i) z[i] = 0.0f;
  return z;
}

// butterfly over 16-lane halves (group-of-32 swizzle, and=0x1f, xor<=8 stays in half)
#define SWZF(x, m) __builtin_bit_cast(float, __builtin_amdgcn_ds_swizzle(__builtin_bit_cast(int, (x)), (0x1f | ((m) << 10))))

// stage one 256x32 f32->bf16 A tile + 32x64 f32->bf16 (transposed) B tile into LDS
__device__ __forceinline__ void stage_f32_tiles(
    const float* __restrict__ X, const float* __restrict__ W,
    int m0, int n0, int kk, int tid,
    unsigned short* __restrict__ Asb, unsigned short* __restrict__ Bsb)
{
#pragma unroll
  for (int j = 0; j < 8; ++j) {
    int idx = tid + 256 * j;            // 2048 float4s
    int off = idx * 4;
    int row = off >> 5, col = off & 31;
    float4 x4 = *(const float4*)(X + (size_t)(m0 + row) * DIMK + kk + col);
    uint2 p;
    p.x = pk2bf(x4.x, x4.y);
    p.y = pk2bf(x4.z, x4.w);
    *(uint2*)(Asb + row * 32 + col) = p;
  }
#pragma unroll
  for (int j = 0; j < 2; ++j) {
    int idx = tid + 256 * j;            // 512 float4s
    int off = idx * 4;
    int kr = off >> 6, col = off & 63;
    float4 w4 = *(const float4*)(W + (size_t)(kk + kr) * HD + n0 + col);
    Bsb[(col + 0) * 32 + kr] = f2bf(w4.x);
    Bsb[(col + 1) * 32 + kr] = f2bf(w4.y);
    Bsb[(col + 2) * 32 + kr] = f2bf(w4.z);
    Bsb[(col + 3) * 32 + kr] = f2bf(w4.w);
  }
}

// ---------------------------------------------------------------------------
// Kernel 1: fused Q/K/V projection. Y = X@W + b, cast bf16, head-scatter store.
// channel c -> head h = c%16, dim d = c/16   (reference "(d h)" split)
// q,k -> [b][h][t][64] row-major ; v -> [b][h][64][t] (transposed)
// 256x64 tile / WG, double-buffered LDS, 8 WMMAs per wave per K-step.
// ---------------------------------------------------------------------------
__global__ __launch_bounds__(256) void qkv_proj_kernel(
    const float* __restrict__ qx, const float* __restrict__ kx, const float* __restrict__ vx,
    const float* __restrict__ Wq, const float* __restrict__ Wk, const float* __restrict__ Wv,
    const float* __restrict__ bq, const float* __restrict__ bk, const float* __restrict__ bv,
    unsigned short* __restrict__ qh, unsigned short* __restrict__ kh,
    unsigned short* __restrict__ vt)
{
  __shared__ unsigned short As[2][256 * 32];  // [m][k] bf16
  __shared__ unsigned short Bs[2][64 * 32];   // [n][k] bf16 (W transposed)

  const int which = blockIdx.z;
  const float* X    = (which == 0) ? qx : (which == 1) ? kx : vx;
  const float* W    = (which == 0) ? Wq : (which == 1) ? Wk : Wv;
  const float* bias = (which == 0) ? bq : (which == 1) ? bk : bv;

  const int m0 = blockIdx.x * 256;
  const int n0 = blockIdx.y * 64;
  const int tid  = threadIdx.x;
  const int lane = tid & 31;
  const int wv   = tid >> 5;
  const int sub  = lane >> 4;
  const int l15  = lane & 15;
  const int koff = sub ? 8 : 0;
  const int kb   = sub ? 8 : 0;

  v8f acc0[4], acc1[4];
#pragma unroll
  for (int i = 0; i < 4; ++i) { acc0[i] = zero8(); acc1[i] = zero8(); }

  stage_f32_tiles(X, W, m0, n0, 0, tid, As[0], Bs[0]);
  __syncthreads();

  for (int kk = 32; kk <= DIMK; kk += 32) {
    const int cur = ((kk >> 5) - 1) & 1;
    const int nxt = cur ^ 1;
    if (kk < DIMK) {
      stage_f32_tiles(X, W, m0, n0, kk, tid, As[nxt], Bs[nxt]);
      __builtin_prefetch(W + (size_t)(kk + (tid >> 3)) * HD + n0, 0, 1);
    }
    const unsigned short* Asb = As[cur];
    const unsigned short* Bsb = Bs[cur];
    const int r0 = wv * 32 + l15;
    v16bf a0 = load_fragA(Asb + r0 * 32 + koff);
    v16bf a1 = load_fragA(Asb + (r0 + 16) * 32 + koff);
    v16bf bfr[4];
#pragma unroll
    for (int i = 0; i < 4; ++i) bfr[i] = load_fragB(Bsb + (16 * i + l15) * 32 + kb);
#pragma unroll
    for (int i = 0; i < 4; ++i) acc0[i] = wmma_bf16(a0, bfr[i], acc0[i]);
#pragma unroll
    for (int i = 0; i < 4; ++i) acc1[i] = wmma_bf16(a1, bfr[i], acc1[i]);
    __syncthreads();
  }

  // epilogue: bias + head scatter
#pragma unroll
  for (int i = 0; i < 4; ++i) {
    int c = n0 + 16 * i + l15;
    float bia = bias[c];
    int h = c & 15, d = c >> 4;
#pragma unroll
    for (int g = 0; g < 2; ++g) {
      v8f av = g ? acc1[i] : acc0[i];
#pragma unroll
      for (int r = 0; r < 8; r += 2) {
        int grow = m0 + wv * 32 + g * 16 + r + 8 * sub;  // even; pair is consecutive rows
        int bb = grow >> 11, t = grow & (TLEN - 1);
        if (which == 2) {
          // transposed V: consecutive t -> contiguous, pack two bf16 per b32 store
          unsigned pv = pk2bf(av[r] + bia, av[r + 1] + bia);
          *(unsigned*)(vt + ((size_t)(bb * HEADS + h) * DH + d) * TLEN + t) = pv;
        } else {
          unsigned short* dst = (which == 0) ? qh : kh;
          dst[((size_t)(bb * HEADS + h) * TLEN + t) * DH + d]     = f2bf(av[r] + bia);
          dst[((size_t)(bb * HEADS + h) * TLEN + t + 1) * DH + d] = f2bf(av[r + 1] + bia);
        }
      }
    }
  }
}

// ---------------------------------------------------------------------------
// Kernel 2: flash attention. One wave per 16-row Q tile, stream 32 keys/iter.
// ---------------------------------------------------------------------------
__global__ __launch_bounds__(256) void attn_kernel(
    const unsigned short* __restrict__ qh,
    const unsigned short* __restrict__ kh,
    const unsigned short* __restrict__ vt,
    unsigned short* __restrict__ aout)
{
  __shared__ unsigned short pbuf[8][16 * 32];  // per-wave P transpose buffer

  const int tid  = threadIdx.x;
  const int lane = tid & 31;
  const int wv   = tid >> 5;
  const int sub  = lane >> 4;
  const int l15  = lane & 15;
  const int h = blockIdx.y, b = blockIdx.z;
  const int q0 = (blockIdx.x * 8 + wv) * 16;

  const unsigned short* Qb = qh + (size_t)(b * HEADS + h) * TLEN * DH;
  const unsigned short* Kb = kh + (size_t)(b * HEADS + h) * TLEN * DH;
  const unsigned short* Vb = vt + (size_t)(b * HEADS + h) * DH * TLEN;

  const int aoff = sub ? 8 : 0;   // A-frag K offset
  const int kb16 = sub ? 8 : 0;   // B-frag K offset

  const unsigned short* qrow = Qb + (size_t)(q0 + l15) * DH;
  v16bf aq0 = load_fragA(qrow + 0  + aoff);
  v16bf aq1 = load_fragA(qrow + 32 + aoff);

  v8f o[4];
  float mrow[8], lrow[8];
#pragma unroll
  for (int i = 0; i < 4; ++i) o[i] = zero8();
#pragma unroll
  for (int r = 0; r < 8; ++r) { mrow[r] = -1e30f; lrow[r] = 0.0f; }

  for (int jt = 0; jt < TLEN; jt += 32) {
    // ---- scores S = Q K^T (16 x 32): batch fragment loads, then 4 WMMAs ----
    const unsigned short* krow0 = Kb + (size_t)(jt + l15) * DH;
    const unsigned short* krow1 = Kb + (size_t)(jt + 16 + l15) * DH;
    v16bf bk0a = load_fragB(krow0 + 0  + kb16);
    v16bf bk0b = load_fragB(krow0 + 32 + kb16);
    v16bf bk1a = load_fragB(krow1 + 0  + kb16);
    v16bf bk1b = load_fragB(krow1 + 32 + kb16);
    v8f s0 = zero8(), s1 = zero8();
    s0 = wmma_bf16(aq0, bk0a, s0);
    s0 = wmma_bf16(aq1, bk0b, s0);
    s1 = wmma_bf16(aq0, bk1a, s1);
    s1 = wmma_bf16(aq1, bk1b, s1);

    // ---- online softmax (row = r + 8*sub, columns across 16-lane half) ----
#pragma unroll
    for (int r = 0; r < 8; ++r) {
      float x0 = s0[r] * SCALE, x1 = s1[r] * SCALE;
      float mx = fmaxf(x0, x1);
      mx = fmaxf(mx, SWZF(mx, 1)); mx = fmaxf(mx, SWZF(mx, 2));
      mx = fmaxf(mx, SWZF(mx, 4)); mx = fmaxf(mx, SWZF(mx, 8));
      float nm  = fmaxf(mrow[r], mx);
      float fac = __expf(mrow[r] - nm);
      float p0 = __expf(x0 - nm), p1 = __expf(x1 - nm);
      float sm = p0 + p1;
      sm += SWZF(sm, 1); sm += SWZF(sm, 2); sm += SWZF(sm, 4); sm += SWZF(sm, 8);
      lrow[r] = lrow[r] * fac + sm;
      mrow[r] = nm;
      o[0][r] *= fac; o[1][r] *= fac; o[2][r] *= fac; o[3][r] *= fac;
      int prow = r + 8 * sub;
      pbuf[wv][prow * 32 + l15]      = f2bf(p0);
      pbuf[wv][prow * 32 + l15 + 16] = f2bf(p1);
    }

    // make P stores visible to the A-fragment reload (same wave, DS in-order)
    asm volatile("s_wait_dscnt 0" ::: "memory");

    // ---- O += P V (16 x 64): batch V fragment loads, then 4 WMMAs ----
    v16bf ap = load_fragA(&pbuf[wv][l15 * 32 + aoff]);
    v16bf bv[4];
#pragma unroll
    for (int i = 0; i < 4; ++i)
      bv[i] = load_fragB(Vb + (size_t)(16 * i + l15) * TLEN + jt + kb16);
#pragma unroll
    for (int i = 0; i < 4; ++i)
      o[i] = wmma_bf16(ap, bv[i], o[i]);
  }

  // finalize: O /= l, store to [b][t][(h d)] channel order (h outer)
#pragma unroll
  for (int i = 0; i < 4; ++i) {
    int c = h * DH + 16 * i + l15;
#pragma unroll
    for (int r = 0; r < 8; ++r) {
      int t = q0 + r + 8 * sub;
      aout[(size_t)(b * TLEN + t) * HD + c] = f2bf(o[i][r] / lrow[r]);
    }
  }
}

// ---------------------------------------------------------------------------
// Kernel 3: output projection. out = attn(bf16) @ W0(f32->bf16) -> f32
// 256x64 tile / WG, double-buffered LDS.
// ---------------------------------------------------------------------------
__global__ __launch_bounds__(256) void out_proj_kernel(
    const unsigned short* __restrict__ A, const float* __restrict__ W0,
    float* __restrict__ out)
{
  __shared__ unsigned short As[2][256 * 32];
  __shared__ unsigned short Bs[2][64 * 32];

  const int m0 = blockIdx.x * 256;
  const int n0 = blockIdx.y * 64;
  const int tid  = threadIdx.x;
  const int lane = tid & 31;
  const int wv   = tid >> 5;
  const int sub  = lane >> 4;
  const int l15  = lane & 15;
  const int koff = sub ? 8 : 0;
  const int kb   = sub ? 8 : 0;

  v8f acc0[4], acc1[4];
#pragma unroll
  for (int i = 0; i < 4; ++i) { acc0[i] = zero8(); acc1[i] = zero8(); }

  // stager for bf16 A tile + f32 B tile
  auto stage = [&](int kk, unsigned short* Asb, unsigned short* Bsb) {
#pragma unroll
    for (int j = 0; j < 4; ++j) {
      int idx = tid + 256 * j;            // 1024 x (8 ushorts)
      int off = idx * 8;
      int row = off >> 5, col = off & 31;
      uint4 a4 = *(const uint4*)(A + (size_t)(m0 + row) * HD + kk + col);
      *(uint4*)(Asb + row * 32 + col) = a4;
    }
#pragma unroll
    for (int j = 0; j < 2; ++j) {
      int idx = tid + 256 * j;
      int off = idx * 4;
      int kr = off >> 6, col = off & 63;
      float4 w4 = *(const float4*)(W0 + (size_t)(kk + kr) * DIMK + n0 + col);
      Bsb[(col + 0) * 32 + kr] = f2bf(w4.x);
      Bsb[(col + 1) * 32 + kr] = f2bf(w4.y);
      Bsb[(col + 2) * 32 + kr] = f2bf(w4.z);
      Bsb[(col + 3) * 32 + kr] = f2bf(w4.w);
    }
  };

  stage(0, As[0], Bs[0]);
  __syncthreads();

  for (int kk = 32; kk <= HD; kk += 32) {
    const int cur = ((kk >> 5) - 1) & 1;
    const int nxt = cur ^ 1;
    if (kk < HD) {
      stage(kk, As[nxt], Bs[nxt]);
      __builtin_prefetch(W0 + (size_t)(kk + (tid >> 3)) * DIMK + n0, 0, 1);
    }
    const unsigned short* Asb = As[cur];
    const unsigned short* Bsb = Bs[cur];
    const int r0 = wv * 32 + l15;
    v16bf a0 = load_fragA(Asb + r0 * 32 + koff);
    v16bf a1 = load_fragA(Asb + (r0 + 16) * 32 + koff);
    v16bf bfr[4];
#pragma unroll
    for (int i = 0; i < 4; ++i) bfr[i] = load_fragB(Bsb + (16 * i + l15) * 32 + kb);
#pragma unroll
    for (int i = 0; i < 4; ++i) acc0[i] = wmma_bf16(a0, bfr[i], acc0[i]);
#pragma unroll
    for (int i = 0; i < 4; ++i) acc1[i] = wmma_bf16(a1, bfr[i], acc1[i]);
    __syncthreads();
  }

#pragma unroll
  for (int i = 0; i < 4; ++i) {
    int c = n0 + 16 * i + l15;
#pragma unroll
    for (int g = 0; g < 2; ++g) {
      v8f av = g ? acc1[i] : acc0[i];
#pragma unroll
      for (int r = 0; r < 8; ++r) {
        int grow = m0 + wv * 32 + g * 16 + r + 8 * sub;
        out[(size_t)grow * DIMK + c] = av[r];
      }
    }
  }
}

// ---------------------------------------------------------------------------
extern "C" void kernel_launch(void* const* d_in, const int* in_sizes, int n_in,
                              void* d_out, int out_size, void* d_ws, size_t ws_size,
                              hipStream_t stream) {
  (void)in_sizes; (void)n_in; (void)out_size; (void)ws_size;
  const size_t headElems = (size_t)BATCH * HEADS * TLEN * DH;  // 4 Mi elems (8 MB bf16)
  unsigned short* qh = (unsigned short*)d_ws;
  unsigned short* kh = qh + headElems;
  unsigned short* vt = kh + headElems;
  unsigned short* ao = vt + headElems;

  qkv_proj_kernel<<<dim3(MTOT / 256, HD / 64, 3), 256, 0, stream>>>(
      (const float*)d_in[0], (const float*)d_in[1], (const float*)d_in[2],
      (const float*)d_in[3], (const float*)d_in[5], (const float*)d_in[7],
      (const float*)d_in[4], (const float*)d_in[6], (const float*)d_in[8],
      qh, kh, vt);

  attn_kernel<<<dim3(TLEN / 128, HEADS, BATCH), 256, 0, stream>>>(qh, kh, vt, ao);

  out_proj_kernel<<<dim3(MTOT / 256, DIMK / 64), 256, 0, stream>>>(
      ao, (const float*)d_in[9], (float*)d_out);
}
